// LSTM_63282048139374
// MI455X (gfx1250) — matrix operand order repaired and checked by reading, not compiled, not run on hardware
//
#include <hip/hip_runtime.h>
#include <math.h>

#define VOCABSZ 50000
#define EMBD    1024
#define HD      512
#define G4      2048      // 4*HD
#define BB      32
#define TT      512
#define MTOT    (BB*TT)   // 16384

typedef float v2f  __attribute__((ext_vector_type(2)));
typedef float v8f  __attribute__((ext_vector_type(8)));
typedef __bf16 v4bf  __attribute__((ext_vector_type(4)));
typedef __bf16 v8bf  __attribute__((ext_vector_type(8)));
typedef __bf16 v16bf __attribute__((ext_vector_type(16)));

__device__ __forceinline__ v8f wmma_f32(v2f a, v2f b, v8f c) {
  return __builtin_amdgcn_wmma_f32_16x16x4_f32(false, a, false, b, (short)0, c, false, false);
}
__device__ __forceinline__ v8f wmma_bf16(v16bf a, v16bf b, v8f c) {
  return __builtin_amdgcn_wmma_f32_16x16x32_bf16(false, a, false, b, (short)0, c, false, false);
}
__device__ __forceinline__ float sigm(float x) { return 1.0f / (1.0f + __expf(-x)); }

// ---------------------------------------------------------------------------
// 1) Embedding lookup: x0[b,t,:] = w_emb[seq[b,t],:]  (row 0 -> zeros)
// ---------------------------------------------------------------------------
__global__ __launch_bounds__(256) void emb_kernel(const int* __restrict__ seq,
                                                  const float* __restrict__ w_emb,
                                                  float* __restrict__ x0) {
  const int bt = blockIdx.x;
  const int tok = seq[bt];
  const float4* src = (const float4*)(w_emb + (size_t)tok * EMBD);
  float4* dst = (float4*)(x0 + (size_t)bt * EMBD);
  float4 z; z.x = z.y = z.z = z.w = 0.f;
  dst[threadIdx.x] = (tok == 0) ? z : src[threadIdx.x];
}

// ---------------------------------------------------------------------------
// 2) fp32 -> bf16 conversion (n multiple of 4)
// ---------------------------------------------------------------------------
__global__ __launch_bounds__(256) void f2bf_kernel(const float* __restrict__ s,
                                                   __bf16* __restrict__ d) {
  const size_t i = ((size_t)blockIdx.x * 256 + threadIdx.x) * 4;
  float4 v = *(const float4*)(s + i);
  v4bf o;
  o[0] = (__bf16)v.x; o[1] = (__bf16)v.y; o[2] = (__bf16)v.z; o[3] = (__bf16)v.w;
  *(v4bf*)(d + i) = o;
}

// ---------------------------------------------------------------------------
// 3) Big GEMM in bf16 (f32 accumulate): Y[m,n] = sum_k X[m,k]*W[n,k] + bias[n]
//    M=16384, N=2048, K=1024.  Wave tile 32x64, wmma 16x16x32 bf16.
// ---------------------------------------------------------------------------
__global__ __launch_bounds__(256) void gemm_gx_bf16_kernel(const __bf16* __restrict__ X,
                                                           const __bf16* __restrict__ W,
                                                           const float* __restrict__ bias,
                                                           float* __restrict__ Y) {
  const int wave = threadIdx.x >> 5;
  const int lane = threadIdx.x & 31;
  const int lr = lane & 15;
  const int lh = lane >> 4;
  const int m0 = blockIdx.x * 64 + (wave >> 2) * 32;
  const int n0 = blockIdx.y * 256 + (wave & 3) * 64;

  v8f acc[2][4] = {};

  // A: lanes 0-15 hold K = 0..15, lanes 16-31 K = 16..31
  const __bf16* xr0 = X + (size_t)(m0 + lr) * EMBD + 16 * lh;
  const __bf16* xr1 = X + (size_t)(m0 + 16 + lr) * EMBD + 16 * lh;
  // B: lanes 0-15 hold K = 0..15, lanes 16-31 K = 16..31 (contiguous 32B)
  const __bf16* wr0 = W + (size_t)(n0 + lr) * EMBD + 16 * lh;
  const __bf16* wr1 = W + (size_t)(n0 + 16 + lr) * EMBD + 16 * lh;
  const __bf16* wr2 = W + (size_t)(n0 + 32 + lr) * EMBD + 16 * lh;
  const __bf16* wr3 = W + (size_t)(n0 + 48 + lr) * EMBD + 16 * lh;

#pragma unroll 2
  for (int k = 0; k < EMBD; k += 32) {
    v8bf a0l = *(const v8bf*)(xr0 + k);
    v8bf a0h = *(const v8bf*)(xr0 + k + 8);
    v8bf a1l = *(const v8bf*)(xr1 + k);
    v8bf a1h = *(const v8bf*)(xr1 + k + 8);
    v16bf a0 = __builtin_shufflevector(a0l, a0h, 0,1,2,3,4,5,6,7,8,9,10,11,12,13,14,15);
    v16bf a1 = __builtin_shufflevector(a1l, a1h, 0,1,2,3,4,5,6,7,8,9,10,11,12,13,14,15);
    v16bf b0 = *(const v16bf*)(wr0 + k);
    v16bf b1 = *(const v16bf*)(wr1 + k);
    v16bf b2 = *(const v16bf*)(wr2 + k);
    v16bf b3 = *(const v16bf*)(wr3 + k);
    acc[0][0] = wmma_bf16(a0, b0, acc[0][0]);
    acc[0][1] = wmma_bf16(a0, b1, acc[0][1]);
    acc[0][2] = wmma_bf16(a0, b2, acc[0][2]);
    acc[0][3] = wmma_bf16(a0, b3, acc[0][3]);
    acc[1][0] = wmma_bf16(a1, b0, acc[1][0]);
    acc[1][1] = wmma_bf16(a1, b1, acc[1][1]);
    acc[1][2] = wmma_bf16(a1, b2, acc[1][2]);
    acc[1][3] = wmma_bf16(a1, b3, acc[1][3]);
  }

#pragma unroll
  for (int ni = 0; ni < 4; ++ni) {
    const int col = n0 + 16 * ni + lr;
    const float bv = bias[col];
#pragma unroll
    for (int mi = 0; mi < 2; ++mi) {
      const int rbase = m0 + 16 * mi + 8 * lh;
      v8f a = acc[mi][ni];
#pragma unroll
      for (int v = 0; v < 8; ++v)
        Y[(size_t)(rbase + v) * G4 + col] = a[v] + bv;
    }
  }
}

// ---------------------------------------------------------------------------
// 4) init: zero h double-buffers and sync counters
// ---------------------------------------------------------------------------
__global__ __launch_bounds__(256) void init_kernel(float* __restrict__ hbuf,
                                                   unsigned int* __restrict__ sync) {
  const int i = blockIdx.x * 256 + threadIdx.x;
  if (i < 2 * 2 * BB * HD) hbuf[i] = 0.f;
  if (i < 2) sync[i] = 0u;
}

// ---------------------------------------------------------------------------
// 5) Persistent recurrent scan (fp32 WMMA to protect the 512-step chain).
//    64 blocks x 256 threads. dir = blockIdx.x/32; WG owns 16 hidden units.
//    Waves split K; i/f/g/o tiles share per-lane (batch,unit) -> local update.
//    gx for the step is prefetched at loop top (latency hidden by the GEMM).
// ---------------------------------------------------------------------------
__global__ __launch_bounds__(256) void lstm_scan_kernel(const float* __restrict__ gxF,
                                                        const float* __restrict__ gxB,
                                                        const float* __restrict__ whhF,
                                                        const float* __restrict__ whhB,
                                                        float* __restrict__ xout,
                                                        float* __restrict__ hbuf,
                                                        unsigned int* __restrict__ sync) {
  __shared__ float red[7 * 2048];   // 56 KB: partials of waves 1..7
  __shared__ float gxs[2048];       //  8 KB: staged gx tile for this step

  const int dir = blockIdx.x >> 5;
  const int wg = blockIdx.x & 31;
  const int u0 = wg * 16;
  const int wave = threadIdx.x >> 5;
  const int lane = threadIdx.x & 31;
  const int lr = lane & 15;
  const int lh = lane >> 4;

  const float* gx = dir ? gxB : gxF;
  const float* whh = dir ? whhB : whhF;
  unsigned int* cnt = sync + dir;
  float* hbase = hbuf + (size_t)dir * 2 * BB * HD;

  float cst[2][8];
#pragma unroll
  for (int mi = 0; mi < 2; ++mi)
#pragma unroll
    for (int v = 0; v < 8; ++v) cst[mi][v] = 0.f;

  const float* wrow[4];
#pragma unroll
  for (int gi = 0; gi < 4; ++gi)
    wrow[gi] = whh + (size_t)(gi * HD + u0 + lr) * HD + wave * 64 + 2 * lh;

  for (int step = 0; step < TT; ++step) {
    const int tIdx = dir ? (TT - 1 - step) : step;
    const float* hr = hbase + (size_t)(step & 1) * BB * HD;
    float* hw = hbase + (size_t)((step + 1) & 1) * BB * HD;

    // ---- prefetch this step's gx tile (2048 values across 256 threads) ----
    float gxr[8];
#pragma unroll
    for (int i = 0; i < 8; ++i) {
      const int j = i * 256 + threadIdx.x;       // j = batch*64 + gi*16 + lr2
      const int lr2 = j & 15, gi2 = (j >> 4) & 3, bb2 = j >> 6;
      gxr[i] = gx[((size_t)bb2 * TT + tIdx) * G4 + gi2 * 512 + u0 + lr2];
    }

    // ---- partial GEMM over this wave's K slice ----
    v8f acc[2][4] = {};
    const float* hr0 = hr + (size_t)lr * HD + wave * 64 + 2 * lh;
    const float* hr1 = hr + (size_t)(16 + lr) * HD + wave * 64 + 2 * lh;
#pragma unroll 4
    for (int kk = 0; kk < 64; kk += 4) {
      v2f a0 = *(const v2f*)(hr0 + kk);
      v2f a1 = *(const v2f*)(hr1 + kk);
#pragma unroll
      for (int gi = 0; gi < 4; ++gi) {
        v2f bf = *(const v2f*)(wrow[gi] + kk);
        acc[0][gi] = wmma_f32(a0, bf, acc[0][gi]);
        acc[1][gi] = wmma_f32(a1, bf, acc[1][gi]);
      }
    }

    // ---- stage gx + partials in LDS ----
#pragma unroll
    for (int i = 0; i < 8; ++i) gxs[i * 256 + threadIdx.x] = gxr[i];
    if (wave != 0) {
#pragma unroll
      for (int mi = 0; mi < 2; ++mi)
#pragma unroll
        for (int gi = 0; gi < 4; ++gi) {
          v8f a = acc[mi][gi];
#pragma unroll
          for (int v = 0; v < 8; ++v)
            red[(wave - 1) * 2048 + (mi * 4 + gi) * 256 + v * 32 + lane] = a[v];
        }
    }
    __syncthreads();

    // ---- wave 0: reduce + cell update ----
    if (wave == 0) {
#pragma unroll
      for (int mi = 0; mi < 2; ++mi) {
#pragma unroll
        for (int v = 0; v < 8; ++v) {
          float gsum[4];
#pragma unroll
          for (int gi = 0; gi < 4; ++gi) {
            float s = acc[mi][gi][v];
#pragma unroll
            for (int w = 0; w < 7; ++w)
              s += red[w * 2048 + (mi * 4 + gi) * 256 + v * 32 + lane];
            gsum[gi] = s;
          }
          const int batch = 16 * mi + 8 * lh + v;
          const int gs = batch * 64 + lr;
          const float gi_ = gsum[0] + gxs[gs];
          const float gf_ = gsum[1] + gxs[gs + 16];
          const float gg_ = gsum[2] + gxs[gs + 32];
          const float go_ = gsum[3] + gxs[gs + 48];
          const float c = sigm(gf_) * cst[mi][v] + sigm(gi_) * tanhf(gg_);
          cst[mi][v] = c;
          const float h = sigm(go_) * tanhf(c);
          hw[batch * HD + u0 + lr] = h;
          xout[((size_t)batch * TT + tIdx) * 1024 + dir * HD + u0 + lr] = h;
        }
      }
    }
    __syncthreads();

    // ---- device-scope barrier over this direction's 32 WGs ----
    if (threadIdx.x == 0) {
      __hip_atomic_fetch_add(cnt, 1u, __ATOMIC_RELEASE, __HIP_MEMORY_SCOPE_AGENT);
      const unsigned int target = 32u * (unsigned int)(step + 1);
      while (__hip_atomic_load(cnt, __ATOMIC_ACQUIRE, __HIP_MEMORY_SCOPE_AGENT) < target)
        __builtin_amdgcn_s_sleep(1);
    }
    __syncthreads();
  }
}

// ---------------------------------------------------------------------------
// 6) Gather hidden state at marker position per row
// ---------------------------------------------------------------------------
__global__ __launch_bounds__(256) void gather_kernel(const int* __restrict__ seq,
                                                     const float* __restrict__ xin,
                                                     float* __restrict__ hidden) {
  __shared__ int sidx;
  const int b = blockIdx.x;
  for (int t = threadIdx.x; t < TT; t += 256)
    if (seq[b * TT + t] == VOCABSZ) sidx = t;
  __syncthreads();
  const int id = sidx;
  for (int k = threadIdx.x; k < 1024; k += 256)
    hidden[b * 1024 + k] = xin[((size_t)b * TT + id) * 1024 + k];
}

// ---------------------------------------------------------------------------
// 7) h1 = tanh(hidden @ w1^T + b1)
// ---------------------------------------------------------------------------
__global__ __launch_bounds__(256) void mlp1_kernel(const float* __restrict__ hidden,
                                                   const float* __restrict__ w1,
                                                   const float* __restrict__ b1,
                                                   float* __restrict__ h1) {
  const int tid = blockIdx.x * 256 + threadIdx.x;
  const int b = tid >> 9;
  const int j = tid & 511;
  const float4* hv = (const float4*)(hidden + (size_t)b * 1024);
  const float4* wr = (const float4*)(w1 + (size_t)j * 1024);
  float s = b1[j];
  for (int k = 0; k < 256; ++k) {
    float4 a = hv[k], w = wr[k];
    s += a.x * w.x + a.y * w.y + a.z * w.z + a.w * w.w;
  }
  h1[b * 512 + j] = tanhf(s);
}

// ---------------------------------------------------------------------------
// 8) logits + BCEWithLogits mean
// ---------------------------------------------------------------------------
__global__ __launch_bounds__(256) void loss_kernel(const float* __restrict__ h1,
                                                   const float* __restrict__ w2,
                                                   const float* __restrict__ b2,
                                                   const float* __restrict__ label,
                                                   float* __restrict__ out) {
  __shared__ float red[32];
  const int tid = threadIdx.x;
  if (tid < 32) {
    float s = b2[0];
    for (int j = 0; j < 512; ++j) s += h1[tid * 512 + j] * w2[j];
    const float z = s, y = label[tid];
    red[tid] = fmaxf(z, 0.f) + log1pf(__expf(-fabsf(z))) - z * y;
  }
  __syncthreads();
  if (tid == 0) {
    float s = 0.f;
    for (int i = 0; i < 32; ++i) s += red[i];
    out[0] = s / 32.0f;
  }
}

// ---------------------------------------------------------------------------
// host launcher
// ---------------------------------------------------------------------------
extern "C" void kernel_launch(void* const* d_in, const int* in_sizes, int n_in,
                              void* d_out, int out_size, void* d_ws, size_t ws_size,
                              hipStream_t stream) {
  (void)in_sizes; (void)n_in; (void)out_size; (void)ws_size;

  const int*   seq   = (const int*)d_in[0];
  const float* label = (const float*)d_in[1];
  const float* w_emb = (const float*)d_in[2];
  const float* w_ih  = (const float*)d_in[3];   // [2][2][2048][1024]
  const float* w_hh  = (const float*)d_in[4];   // [2][2][2048][512]
  const float* bias  = (const float*)d_in[5];   // [2][2][2048]
  const float* w1    = (const float*)d_in[6];
  const float* b1    = (const float*)d_in[7];
  const float* w2    = (const float*)d_in[8];
  const float* b2    = (const float*)d_in[9];
  float* out = (float*)d_out;

  char* ws = (char*)d_ws;
  constexpr size_t SZ_X   = (size_t)MTOT * EMBD * sizeof(float);     // 64 MiB
  constexpr size_t SZ_GX  = (size_t)MTOT * G4 * sizeof(float);       // 128 MiB
  constexpr size_t OFF_X0   = 0;
  constexpr size_t OFF_X1   = OFF_X0 + SZ_X;
  constexpr size_t OFF_GXF  = OFF_X1 + SZ_X;
  constexpr size_t OFF_GXB  = OFF_GXF + SZ_GX;
  constexpr size_t OFF_H    = OFF_GXB + SZ_GX;
  constexpr size_t OFF_SYNC = OFF_H + (size_t)2 * 2 * BB * HD * sizeof(float);
  constexpr size_t OFF_HID  = OFF_SYNC + 256;
  constexpr size_t OFF_H1   = OFF_HID + (size_t)BB * 1024 * sizeof(float);
  constexpr size_t OFF_XB   = OFF_H1 + (size_t)BB * HD * sizeof(float);
  constexpr size_t OFF_WB   = OFF_XB + (size_t)MTOT * EMBD * sizeof(__bf16);

  float* x0  = (float*)(ws + OFF_X0);
  float* x1  = (float*)(ws + OFF_X1);
  float* gxF = (float*)(ws + OFF_GXF);
  float* gxB = (float*)(ws + OFF_GXB);
  float* hbf = (float*)(ws + OFF_H);
  unsigned int* sync = (unsigned int*)(ws + OFF_SYNC);
  float* hid = (float*)(ws + OFF_HID);
  float* h1  = (float*)(ws + OFF_H1);
  __bf16* xb = (__bf16*)(ws + OFF_XB);
  __bf16* wb = (__bf16*)(ws + OFF_WB);   // [2][2][2048][1024] bf16

  // convert all w_ih to bf16 once (8.4M elems, 4/thread)
  f2bf_kernel<<<(4 * G4 * EMBD) / 1024, 256, 0, stream>>>(w_ih, wb);

  emb_kernel<<<MTOT, 256, 0, stream>>>(seq, w_emb, x0);

  for (int l = 0; l < 2; ++l) {
    const float* xin = l ? x1 : x0;
    float*       xo  = l ? x0 : x1;
    const __bf16* wbF = wb + (size_t)(l * 2 + 0) * G4 * EMBD;
    const __bf16* wbB = wb + (size_t)(l * 2 + 1) * G4 * EMBD;
    const float* whhF = w_hh + (size_t)(l * 2 + 0) * G4 * HD;
    const float* whhB = w_hh + (size_t)(l * 2 + 1) * G4 * HD;
    const float* bF = bias + (size_t)(l * 2 + 0) * G4;
    const float* bB = bias + (size_t)(l * 2 + 1) * G4;

    f2bf_kernel<<<((size_t)MTOT * EMBD) / 1024, 256, 0, stream>>>(xin, xb);
    gemm_gx_bf16_kernel<<<dim3(MTOT / 64, G4 / 256), 256, 0, stream>>>(xb, wbF, bF, gxF);
    gemm_gx_bf16_kernel<<<dim3(MTOT / 64, G4 / 256), 256, 0, stream>>>(xb, wbB, bB, gxB);
    init_kernel<<<256, 256, 0, stream>>>(hbf, sync);
    lstm_scan_kernel<<<64, 256, 0, stream>>>(gxF, gxB, whhF, whhB, xo, hbf, sync);
  }

  gather_kernel<<<BB, 256, 0, stream>>>(seq, x0, hid);   // layer 1 wrote x0
  mlp1_kernel<<<64, 256, 0, stream>>>(hid, w1, b1, h1);
  loss_kernel<<<1, 256, 0, stream>>>(h1, w2, b2, label, out);
}
